// DecoderRNN_11716670783566
// MI455X (gfx1250) — matrix operand order repaired and checked by reading, not compiled, not run on hardware
//
#include <hip/hip_runtime.h>
#include <math.h>

typedef __bf16 bf16;
typedef __attribute__((ext_vector_type(16))) __bf16 v16bf;
typedef __attribute__((ext_vector_type(8)))  __bf16 v8bf;
typedef __attribute__((ext_vector_type(8)))  float  v8f;

#define T_STEPS 40
#define BATCH   512
#define DIN     2048
#define EMBD    200
#define HID     1024
#define NCLS    151
#define KCAT    (DIN + EMBD + HID)   // 3272 = real K of fused GEMM
#define KPAD    3328                 // padded to multiple of 64
#define NGATES  (6 * HID)            // 6144
#define NOUTPAD 160                  // 151 padded to multiple of 16

union Frag { v16bf v; v8bf h[2]; };

// ---------------------------------------------------------------------------
// bf16 WMMA GEMM:  Cd[m][n] = sum_k A[m][k] * W[n][k] + bias[n]
// A row-major [M x K] (stride lda), W row-major [N x K] (stride ldw).
// Each wave computes a 32x(16*NTIL) output tile: 2 M-tiles x NTIL N-tiles.
// B-fragments are reused across both M-tiles; A-fragments across NTIL tiles.
// Launcher guarantees exact wave count and nTiles == nGroups*NTIL, K%32==0,
// so control flow is fully wave-uniform => EXEC all-ones at every WMMA.
// ---------------------------------------------------------------------------
template <int NTIL>
__global__ void wmma_gemm_kernel(const bf16* __restrict__ A, int lda,
                                 const bf16* __restrict__ W, int ldw,
                                 float* __restrict__ Cd, int ldc,
                                 const float* __restrict__ bias,
                                 int nGroups, int K, int nLimit)
{
    const int lane = threadIdx.x & 31;
    const int wid  = (blockIdx.x * blockDim.x + threadIdx.x) >> 5;
    const int mp   = wid / nGroups;  // M-pair index (2 tiles = 32 rows)
    const int ng   = wid % nGroups;
    const int hi   = lane >> 4;      // half-wave selector
    const int l16  = lane & 15;

    v8f acc[2][NTIL];
    #pragma unroll
    for (int mi = 0; mi < 2; ++mi)
        #pragma unroll
        for (int i = 0; i < NTIL; ++i)
            acc[mi][i] = (v8f){0.f, 0.f, 0.f, 0.f, 0.f, 0.f, 0.f, 0.f};

    // A-fragment (16-bit A 16x32 layout, ISA 7.12.2):
    //   lanes 0-15 : M=lane,    K = {0..7, 16..23}
    //   lanes 16-31: M=lane-16, K = {8..15, 24..31}
    const bf16* aRow0 = A + (size_t)(mp * 32 + l16) * lda + hi * 8;
    const bf16* aRow1 = aRow0 + (size_t)16 * lda;

    // B-fragment (32x16): lane l holds column N=l16, same K order as A
    // (lanes 0-15: K={0..7,16..23}, lanes 16-31: K={8..15,24..31}).
    const bf16* wRow[NTIL];
    #pragma unroll
    for (int nt = 0; nt < NTIL; ++nt)
        wRow[nt] = W + (size_t)((ng * NTIL + nt) * 16 + l16) * ldw + hi * 8;

    for (int k0 = 0; k0 < K; k0 += 32) {
        // Issue ALL loads for this K-slice first, then a burst of 2*NTIL WMMAs.
        Frag a0, a1;
        a0.h[0] = *(const v8bf*)(aRow0 + k0);
        a0.h[1] = *(const v8bf*)(aRow0 + k0 + 16);
        a1.h[0] = *(const v8bf*)(aRow1 + k0);
        a1.h[1] = *(const v8bf*)(aRow1 + k0 + 16);

        Frag b[NTIL];
        #pragma unroll
        for (int nt = 0; nt < NTIL; ++nt) {
            b[nt].h[0] = *(const v8bf*)(wRow[nt] + k0);
            b[nt].h[1] = *(const v8bf*)(wRow[nt] + k0 + 16);
        }
        __builtin_prefetch(aRow0 + k0 + 32, 0, 3);   // global_prefetch_b8

        #pragma unroll
        for (int nt = 0; nt < NTIL; ++nt)
            acc[0][nt] = __builtin_amdgcn_wmma_f32_16x16x32_bf16(
                false, a0.v, false, b[nt].v, (short)0, acc[0][nt], false, false);
        #pragma unroll
        for (int nt = 0; nt < NTIL; ++nt)
            acc[1][nt] = __builtin_amdgcn_wmma_f32_16x16x32_bf16(
                false, a1.v, false, b[nt].v, (short)0, acc[1][nt], false, false);
    }

    // C/D layout: VGPR r -> M = r + 8*hi, N = l16
    #pragma unroll
    for (int mi = 0; mi < 2; ++mi) {
        #pragma unroll
        for (int nt = 0; nt < NTIL; ++nt) {
            const int n = (ng * NTIL + nt) * 16 + l16;
            if (n >= nLimit) continue;             // per-lane store mask only
            const float bv = bias ? bias[n] : 0.f;
            #pragma unroll
            for (int r = 0; r < 8; ++r) {
                const int m = mp * 32 + mi * 16 + r + hi * 8;
                Cd[(size_t)m * ldc + n] = acc[mi][nt][r] + bv;
            }
        }
    }
}

// ---------------------------------------------------------------------------
// Weight preparation: fused bf16 weight matrix Wc [6H x KPAD]:
//   cols 0..2247    = W_in row (x-part then embed-part, contiguous)
//   cols 2248..3271 = W_st (rows < 5H) or 0 (highway row-block)
//   cols 3272..     = 0 padding
// ---------------------------------------------------------------------------
__global__ void prep_wc_kernel(const float* __restrict__ W_in,
                               const float* __restrict__ W_st,
                               bf16* __restrict__ Wc)
{
    const long total  = (long)NGATES * KPAD;
    const long stride = (long)gridDim.x * blockDim.x;
    for (long i = (long)blockIdx.x * blockDim.x + threadIdx.x; i < total; i += stride) {
        const int r = (int)(i / KPAD);
        const int k = (int)(i % KPAD);
        float v;
        if (k < DIN + EMBD)      v = W_in[(size_t)r * (DIN + EMBD) + k];
        else if (k < KCAT)       v = (r < 5 * HID) ? W_st[(size_t)r * HID + (k - (DIN + EMBD))] : 0.f;
        else                     v = 0.f;
        Wc[i] = (bf16)v;
    }
}

__global__ void prep_bias_kernel(const float* __restrict__ b_in,
                                 const float* __restrict__ b_st,
                                 float* __restrict__ bc)
{
    const int r = blockIdx.x * blockDim.x + threadIdx.x;
    if (r < NGATES) bc[r] = b_in[r] + ((r < 5 * HID) ? b_st[r] : 0.f);
}

__global__ void prep_wout_kernel(const float* __restrict__ W_out,
                                 bf16* __restrict__ Wob)
{
    const int i = blockIdx.x * blockDim.x + threadIdx.x;
    if (i >= NOUTPAD * HID) return;
    const int n = i / HID, k = i % HID;
    Wob[i] = (bf16)((n < NCLS) ? W_out[(size_t)n * HID + k] : 0.f);
}

// ---------------------------------------------------------------------------
// Initial activation buffer A[b][k] = [ x_0 | embed_w[0] | h=0 | pad ] (bf16),
// and zero fp32 cell state.
// ---------------------------------------------------------------------------
__global__ void init_kernel(const float* __restrict__ seq,
                            const float* __restrict__ embed_w,
                            bf16* __restrict__ Abuf,
                            float* __restrict__ cmem)
{
    const long stride = (long)gridDim.x * blockDim.x;
    const long totalA = (long)BATCH * KPAD;
    for (long i = (long)blockIdx.x * blockDim.x + threadIdx.x; i < totalA; i += stride) {
        const int b = (int)(i / KPAD);
        const int k = (int)(i % KPAD);
        float v;
        if (k < DIN)             v = seq[(size_t)b * DIN + k];      // timestep 0
        else if (k < DIN + EMBD) v = embed_w[k - DIN];              // 'start' embed
        else                     v = 0.f;                           // h0 + pad
        Abuf[i] = (bf16)v;
    }
    const long totalC = (long)BATCH * HID;
    for (long i = (long)blockIdx.x * blockDim.x + threadIdx.x; i < totalC; i += stride)
        cmem[i] = 0.f;
}

// ---------------------------------------------------------------------------
// LSTM gates + highway. C1 already contains b_in + b_st and the state GEMM.
// ---------------------------------------------------------------------------
__device__ __forceinline__ float sigm(float x) { return 1.f / (1.f + expf(-x)); }

__global__ void gates_kernel(const float* __restrict__ C1,
                             float* __restrict__ cmem,
                             bf16* __restrict__ Abuf)
{
    const int idx = blockIdx.x * blockDim.x + threadIdx.x;
    if (idx >= BATCH * HID) return;
    const int b = idx >> 10;
    const int j = idx & (HID - 1);
    const float* row = C1 + (size_t)b * NGATES;

    const float ig = sigm(row[j]);
    const float fg = sigm(row[HID + j]);
    const float mi = tanhf(row[2 * HID + j]);
    const float og = sigm(row[3 * HID + j]);
    const float hg = sigm(row[4 * HID + j]);
    const float p5 = row[5 * HID + j];

    const float cm  = ig * mi + fg * cmem[idx];
    float out = og * tanhf(cm);
    out = hg * out + (1.f - hg) * p5;

    cmem[idx] = cm;
    Abuf[(size_t)b * KPAD + (DIN + EMBD) + j] = (bf16)out;
}

// ---------------------------------------------------------------------------
// Per-row (one wave32 per batch row): argmax over pred[1:151] with
// first-occurrence tie-break, label feedback, commit write (raw int32),
// embedding gather -> bf16 A columns, next-x convert -> bf16 A columns.
// ---------------------------------------------------------------------------
__global__ void feedback_kernel(int t,
                                const float* __restrict__ dists,
                                int* __restrict__ commits,
                                const int* __restrict__ labels,
                                const float* __restrict__ embed_w,
                                const float* __restrict__ seq,
                                bf16* __restrict__ Abuf)
{
    const int wid  = (blockIdx.x * blockDim.x + threadIdx.x) >> 5;
    const int lane = threadIdx.x & 31;
    if (wid >= BATCH) return;

    const int row = t * BATCH + wid;
    const float* pred = dists + (size_t)row * NCLS;

    float bv = -3.402823e38f;
    int   bi = NCLS;
    for (int n = 1 + lane; n < NCLS; n += 32) {
        const float v = pred[n];
        if (v > bv) { bv = v; bi = n; }       // strict > keeps first max in-lane
    }
    for (int off = 16; off > 0; off >>= 1) {
        const float ov = __shfl_down(bv, off);
        const int   oi = __shfl_down(bi, off);
        if (ov > bv || (ov == bv && oi < bi)) { bv = ov; bi = oi; }
    }
    bi = __shfl(bi, 0);

    const int lab  = labels[row];
    const int lab2 = (lab == 0) ? bi : lab;
    if (lane == 0) commits[row] = lab2;

    bf16* arow = Abuf + (size_t)wid * KPAD;
    const float* e = embed_w + (size_t)(lab2 + 1) * EMBD;
    for (int i = lane; i < EMBD; i += 32) arow[DIN + i] = (bf16)e[i];

    if (t + 1 < T_STEPS) {
        const float* x = seq + ((size_t)(t + 1) * BATCH + wid) * DIN;
        for (int i = lane; i < DIN; i += 32) arow[i] = (bf16)x[i];
    }
}

// ---------------------------------------------------------------------------
// Workspace layout (all offsets 256-byte aligned)
// ---------------------------------------------------------------------------
#define OFF_WC   ((size_t)0)                                       // 40,894,464 B
#define OFF_BC   (OFF_WC   + (size_t)NGATES * KPAD * 2)            //     24,576 B
#define OFF_WOB  (OFF_BC   + (size_t)NGATES * 4)                   //    327,680 B
#define OFF_A    (OFF_WOB  + (size_t)NOUTPAD * HID * 2)            //  3,407,872 B
#define OFF_C1   (OFF_A    + (size_t)BATCH * KPAD * 2)             // 12,582,912 B
#define OFF_CM   (OFF_C1   + (size_t)BATCH * NGATES * 4)           //  2,097,152 B

extern "C" void kernel_launch(void* const* d_in, const int* in_sizes, int n_in,
                              void* d_out, int out_size, void* d_ws, size_t ws_size,
                              hipStream_t stream)
{
    const float* seq     = (const float*)d_in[0];
    const int*   labels  = (const int*)  d_in[1];
    const float* W_in    = (const float*)d_in[2];
    const float* b_in    = (const float*)d_in[3];
    const float* W_st    = (const float*)d_in[4];
    const float* b_st    = (const float*)d_in[5];
    const float* W_out   = (const float*)d_in[6];
    const float* b_out   = (const float*)d_in[7];
    const float* embed_w = (const float*)d_in[8];

    char*  ws   = (char*)d_ws;
    bf16*  Wc   = (bf16*) (ws + OFF_WC);
    float* bc   = (float*)(ws + OFF_BC);
    bf16*  Wob  = (bf16*) (ws + OFF_WOB);
    bf16*  Abuf = (bf16*) (ws + OFF_A);
    float* C1   = (float*)(ws + OFF_C1);
    float* cmem = (float*)(ws + OFF_CM);

    float* dists   = (float*)d_out;
    int*   commits = (int*)((float*)d_out + (size_t)T_STEPS * BATCH * NCLS);

    // One-time (per launch) weight/activation prep — deterministic.
    prep_wc_kernel  <<<2048, 256, 0, stream>>>(W_in, W_st, Wc);
    prep_bias_kernel<<<(NGATES + 255) / 256, 256, 0, stream>>>(b_in, b_st, bc);
    prep_wout_kernel<<<(NOUTPAD * HID + 255) / 256, 256, 0, stream>>>(W_out, Wob);
    init_kernel     <<<1024, 256, 0, stream>>>(seq, embed_w, Abuf, cmem);

    // Main GEMM: 16 m-pairs * (384 n-tiles / 4) = 1536 waves = 192 blocks.
    const int nGroupsMain = (NGATES / 16) / 4;          // 96
    const int blocksMain  = (16 * nGroupsMain * 32) / 256;
    // Output GEMM: 16 m-pairs * (10 n-tiles / 5) = 32 waves = 4 blocks.
    const int nGroupsOut  = (NOUTPAD / 16) / 5;         // 2
    const int blocksOut   = (16 * nGroupsOut * 32) / 256;

    for (int t = 0; t < T_STEPS; ++t) {
        // pre-activations for all 6 gate blocks (x, embed, state fused):
        wmma_gemm_kernel<4><<<blocksMain, 256, 0, stream>>>(
            Abuf, KPAD, Wc, KPAD, C1, NGATES, bc,
            nGroupsMain, KPAD, NGATES);

        gates_kernel<<<(BATCH * HID + 255) / 256, 256, 0, stream>>>(C1, cmem, Abuf);

        // pred_t = h_t @ W_out^T + b_out, streamed straight into d_out:
        wmma_gemm_kernel<5><<<blocksOut, 256, 0, stream>>>(
            Abuf + (DIN + EMBD), KPAD, Wob, HID,
            dists + (size_t)t * BATCH * NCLS, NCLS, b_out,
            nGroupsOut, HID, NCLS);

        feedback_kernel<<<(BATCH * 32 + 255) / 256, 256, 0, stream>>>(
            t, dists, commits, labels, embed_w, seq, Abuf);
    }
}
